// MultiHeadAttention_34445637714601
// MI455X (gfx1250) — matrix operand order repaired and checked by reading, not compiled, not run on hardware
//
#include <hip/hip_runtime.h>
#include <math.h>

#define BB  4
#define SS  2048
#define DIMM 1024
#define HH  16
#define DHH 64

typedef __bf16 bf16;
typedef __attribute__((ext_vector_type(16))) __bf16 v16bf;
typedef __attribute__((ext_vector_type(8)))  __bf16 v8bf;
typedef __attribute__((ext_vector_type(8)))  float  v8f;

// ---------------- helpers ----------------
static __device__ inline bf16 f2bf(float f) {
    unsigned u = __builtin_bit_cast(unsigned, f);
    unsigned r = u + 0x7FFFu + ((u >> 16) & 1u);      // round-to-nearest-even
    unsigned short s = (unsigned short)(r >> 16);
    return __builtin_bit_cast(bf16, s);
}

// A-fragment 16x32 bf16 (MxK), row-major source with row stride ldr (elements).
// Lane layout (ISA 7.12.2): lanes 0-15: M=lane, K={0..7,16..23}; lanes 16-31: M=lane-16, K={8..15,24..31}
static __device__ inline v16bf load_a16x32(const bf16* __restrict__ base, int ldr, int lane) {
    int half = lane >> 4, m = lane & 15;
    const bf16* p = base + m * ldr + half * 8;
    v8bf lo = *(const v8bf*)(p);
    v8bf hi = *(const v8bf*)(p + 16);
    return __builtin_shufflevector(lo, hi, 0,1,2,3,4,5,6,7,8,9,10,11,12,13,14,15);
}

// B-fragment 32x16 bf16 (KxN) where the source stores B TRANSPOSED: Bt[n][k], row stride ldr.
// Lane layout: lane<16: N=lane, K={0..7,16..23}; lane>=16: N=lane-16, K={8..15,24..31}.
static __device__ inline v16bf load_b32x16T(const bf16* __restrict__ baseT, int ldr, int lane) {
    int half = lane >> 4, n = lane & 15;
    const bf16* p = baseT + n * ldr + half * 8;
    v8bf lo = *(const v8bf*)(p);
    v8bf hi = *(const v8bf*)(p + 16);
    return __builtin_shufflevector(lo, hi, 0,1,2,3,4,5,6,7,8,9,10,11,12,13,14,15);
}

static __device__ inline v8f wmma_bf16(v16bf a, v16bf b, v8f c) {
    return __builtin_amdgcn_wmma_f32_16x16x32_bf16(false, a, false, b, (short)0, c, false, false);
}

static __device__ inline float rmax16(float v) {
    #pragma unroll
    for (int o = 1; o < 16; o <<= 1) v = fmaxf(v, __shfl_xor(v, o, 32));
    return v;
}
static __device__ inline float rsum16(float v) {
    #pragma unroll
    for (int o = 1; o < 16; o <<= 1) v += __shfl_xor(v, o, 32);
    return v;
}

// ---------------- conversion kernels ----------------
__global__ void k_cvt_bf16(const float* __restrict__ in, bf16* __restrict__ out, int n) {
    int i = blockIdx.x * blockDim.x + threadIdx.x;
    if (i < n) out[i] = f2bf(in[i]);
}

// W [K,N] row-major -> Wt [N,K] bf16 row-major
__global__ void k_cvt_transpose(const float* __restrict__ W, bf16* __restrict__ Wt, int K, int N) {
    int i = blockIdx.x * blockDim.x + threadIdx.x;
    if (i < K * N) {
        int k = i / N, n = i % N;
        Wt[n * K + k] = f2bf(W[i]);
    }
}

__global__ void k_zero1(float* p) { p[0] = 0.f; }

__global__ void k_final(const float* __restrict__ ent, float* __restrict__ out_ent) {
    out_ent[0] = ent[0] / (float)(BB * HH * SS);
}

// ---------------- QKV projection GEMM (32x64 tile per wave) ----------------
__global__ __launch_bounds__(256) void k_gemm_qkv(
    const bf16* __restrict__ A, const bf16* __restrict__ Bt,
    const float* __restrict__ bias, const float* __restrict__ scale_param,
    bf16* __restrict__ Qo, bf16* __restrict__ Ko, bf16* __restrict__ Vt)
{
    const int N = 3 * DIMM;
    int wave = threadIdx.x >> 5, lane = threadIdx.x & 31;
    int tile = blockIdx.x * 8 + wave;
    int tilesN = N / 64;                     // 48
    int tm = tile / tilesN, tn = tile % tilesN;

    const bf16* ar0 = A + (tm * 32 + 0)  * DIMM;
    const bf16* ar1 = A + (tm * 32 + 16) * DIMM;
    const bf16* br0 = Bt + (tn * 64 + 0)  * DIMM;
    const bf16* br1 = Bt + (tn * 64 + 16) * DIMM;
    const bf16* br2 = Bt + (tn * 64 + 32) * DIMM;
    const bf16* br3 = Bt + (tn * 64 + 48) * DIMM;

    v8f c00 = {}, c01 = {}, c02 = {}, c03 = {};
    v8f c10 = {}, c11 = {}, c12 = {}, c13 = {};

    #pragma unroll 2
    for (int k0 = 0; k0 < DIMM; k0 += 32) {
        __builtin_prefetch(ar0 + k0 + 512, 0, 3);
        __builtin_prefetch(br0 + k0 + 512, 0, 3);
        v16bf a0 = load_a16x32(ar0 + k0, DIMM, lane);
        v16bf a1 = load_a16x32(ar1 + k0, DIMM, lane);
        v16bf b0 = load_b32x16T(br0 + k0, DIMM, lane);
        v16bf b1 = load_b32x16T(br1 + k0, DIMM, lane);
        v16bf b2 = load_b32x16T(br2 + k0, DIMM, lane);
        v16bf b3 = load_b32x16T(br3 + k0, DIMM, lane);
        c00 = wmma_bf16(a0, b0, c00);  c01 = wmma_bf16(a0, b1, c01);
        c02 = wmma_bf16(a0, b2, c02);  c03 = wmma_bf16(a0, b3, c03);
        c10 = wmma_bf16(a1, b0, c10);  c11 = wmma_bf16(a1, b1, c11);
        c12 = wmma_bf16(a1, b2, c12);  c13 = wmma_bf16(a1, b3, c13);
    }

    int half = lane >> 4, n = lane & 15;
    v8f* accs[2][4] = { {&c00,&c01,&c02,&c03}, {&c10,&c11,&c12,&c13} };

    #pragma unroll
    for (int nt = 0; nt < 4; ++nt) {
        int col = tn * 64 + nt * 16 + n;
        int h = col / (3 * DHH);
        int rem = col % (3 * DHH);
        int which = rem / DHH;
        int d = rem % DHH;
        float qs = scale_param[h] * logf((float)SS) * (1.f / 8.f);  // * ln(S) / sqrt(DH)
        float bv = bias[col];
        #pragma unroll
        for (int mi = 0; mi < 2; ++mi) {
            v8f& acc = *accs[mi][nt];
            #pragma unroll
            for (int i = 0; i < 8; ++i) {
                int row = tm * 32 + mi * 16 + i + 8 * half;
                int b_ = row / SS, s_ = row % SS;
                int bh = b_ * HH + h;
                float v = acc[i] + bv;
                if (which == 0)      Qo[(bh * SS + s_) * DHH + d] = f2bf(v * qs);
                else if (which == 1) Ko[(bh * SS + s_) * DHH + d] = f2bf(v);
                else                 Vt[(bh * DHH + d) * SS + s_] = f2bf(v);
            }
        }
    }
}

// ---------------- flash attention (32 queries per wave, 32 keys per step) ----------------
__global__ __launch_bounds__(256) void k_attn(
    const bf16* __restrict__ Q, const bf16* __restrict__ K, const bf16* __restrict__ Vt,
    bf16* __restrict__ O, float* __restrict__ entAcc)
{
    __shared__ __align__(16) bf16 pshm[8][2][16 * 40];   // per wave: two 16x32 P tiles

    int wave = threadIdx.x >> 5, lane = threadIdx.x & 31;
    int half = lane >> 4, n = lane & 15;
    int tile = blockIdx.x * 8 + wave;
    const int qtiles = SS / 32;              // 64
    int bh = tile / qtiles, qt = tile % qtiles;

    const bf16* q  = Q  + (bh * SS + qt * 32) * DHH;
    const bf16* kk = K  + bh * SS * DHH;
    const bf16* vt = Vt + bh * DHH * SS;
    bf16* pw0 = &pshm[wave][0][0];
    bf16* pw1 = &pshm[wave][1][0];

    v16bf aq00 = load_a16x32(q,                DHH, lane);
    v16bf aq01 = load_a16x32(q + 32,           DHH, lane);
    v16bf aq10 = load_a16x32(q + 16*DHH,       DHH, lane);
    v16bf aq11 = load_a16x32(q + 16*DHH + 32,  DHH, lane);

    float mrow0[8], lrow0[8], trow0[8], mrow1[8], lrow1[8], trow1[8];
    #pragma unroll
    for (int i = 0; i < 8; ++i) {
        mrow0[i] = -INFINITY; lrow0[i] = 0.f; trow0[i] = 0.f;
        mrow1[i] = -INFINITY; lrow1[i] = 0.f; trow1[i] = 0.f;
    }
    v8f o00 = {}, o01 = {}, o02 = {}, o03 = {};
    v8f o10 = {}, o11 = {}, o12 = {}, o13 = {};

    // causal mask for a score tile: query sub-block mi, key tile index kt
    auto mask_tile = [&](v8f& s, int mi, int kt) {
        if (kt >= 2 * qt + mi) {
            #pragma unroll
            for (int i = 0; i < 8; ++i) {
                int qrow = qt * 32 + mi * 16 + i + 8 * half;
                if (kt * 16 + n > qrow) s[i] = -INFINITY;
            }
        }
    };
    // online softmax update + P staging for one query sub-block
    auto softmax_update = [&](v8f& sA, v8f& sB, float* mrow, float* lrow, float* trow,
                              v8f& a0, v8f& a1, v8f& a2, v8f& a3, bf16* pw) {
        #pragma unroll
        for (int i = 0; i < 8; ++i) {
            float mx = rmax16(fmaxf(sA[i], sB[i]));
            float mn = fmaxf(mrow[i], mx);
            float al = __expf(mrow[i] - mn);
            float p0 = __expf(sA[i] - mn);
            float p1 = __expf(sB[i] - mn);
            float rs = rsum16(p0 + p1);
            float tc = rsum16((p0 > 0.f ? p0 * sA[i] : 0.f) + (p1 > 0.f ? p1 * sB[i] : 0.f));
            lrow[i] = lrow[i] * al + rs;
            trow[i] = trow[i] * al + tc;
            mrow[i] = mn;
            a0[i] *= al; a1[i] *= al; a2[i] *= al; a3[i] *= al;
            int r = i + 8 * half;
            pw[r * 40 + n]      = f2bf(p0);
            pw[r * 40 + 16 + n] = f2bf(p1);
        }
    };

    int nsteps = qt + 1;                     // key tiles 0 .. 2*qt+1 (16-wide), in pairs
    for (int st = 0; st < nsteps; ++st) {
        int kt0 = 2 * st, kt1 = 2 * st + 1;
        int k0 = kt0 * 16;

        __builtin_prefetch(kk + (k0 + 32) * DHH, 0, 3);   // next key tile

        // ---- scores: both query sub-blocks share the K fragments ----
        v16bf bk0a = load_b32x16T(kk + k0 * DHH,             DHH, lane);
        v16bf bk0b = load_b32x16T(kk + k0 * DHH + 32,        DHH, lane);
        v16bf bk1a = load_b32x16T(kk + (k0 + 16) * DHH,      DHH, lane);
        v16bf bk1b = load_b32x16T(kk + (k0 + 16) * DHH + 32, DHH, lane);
        v8f s00 = {}, s01 = {}, s10 = {}, s11 = {};
        s00 = wmma_bf16(aq00, bk0a, s00);  s00 = wmma_bf16(aq01, bk0b, s00);
        s01 = wmma_bf16(aq00, bk1a, s01);  s01 = wmma_bf16(aq01, bk1b, s01);
        s10 = wmma_bf16(aq10, bk0a, s10);  s10 = wmma_bf16(aq11, bk0b, s10);
        s11 = wmma_bf16(aq10, bk1a, s11);  s11 = wmma_bf16(aq11, bk1b, s11);

        mask_tile(s00, 0, kt0);  mask_tile(s01, 0, kt1);
        mask_tile(s10, 1, kt0);  mask_tile(s11, 1, kt1);

        softmax_update(s00, s01, mrow0, lrow0, trow0, o00, o01, o02, o03, pw0);
        softmax_update(s10, s11, mrow1, lrow1, trow1, o10, o11, o12, o13, pw1);

        // ---- O += P @ V : both sub-blocks share the V fragments ----
        v16bf ap0 = load_a16x32(pw0, 40, lane);
        v16bf ap1 = load_a16x32(pw1, 40, lane);
        v16bf bv0 = load_b32x16T(vt + (0 * 16) * SS + k0, SS, lane);
        v16bf bv1 = load_b32x16T(vt + (1 * 16) * SS + k0, SS, lane);
        v16bf bv2 = load_b32x16T(vt + (2 * 16) * SS + k0, SS, lane);
        v16bf bv3 = load_b32x16T(vt + (3 * 16) * SS + k0, SS, lane);
        o00 = wmma_bf16(ap0, bv0, o00);  o01 = wmma_bf16(ap0, bv1, o01);
        o02 = wmma_bf16(ap0, bv2, o02);  o03 = wmma_bf16(ap0, bv3, o03);
        o10 = wmma_bf16(ap1, bv0, o10);  o11 = wmma_bf16(ap1, bv1, o11);
        o12 = wmma_bf16(ap1, bv2, o12);  o13 = wmma_bf16(ap1, bv3, o13);
    }

    // ---- epilogue: normalize, store O (layout [B,S,H*DH]), entropy ----
    int b_ = bh / HH, h = bh % HH;
    float entSum = 0.f;
    auto epilogue = [&](v8f& a0, v8f& a1, v8f& a2, v8f& a3,
                        float* mrow, float* lrow, float* trow, int mi) {
        #pragma unroll
        for (int i = 0; i < 8; ++i) {
            float inv = 1.f / lrow[i];
            a0[i] *= inv; a1[i] *= inv; a2[i] *= inv; a3[i] *= inv;
            entSum += mrow[i] + logf(lrow[i]) - trow[i] * inv;
            int gr = b_ * SS + qt * 32 + mi * 16 + i + 8 * half;
            int cbase = h * DHH + n;
            O[gr * DIMM + cbase + 0]  = f2bf(a0[i]);
            O[gr * DIMM + cbase + 16] = f2bf(a1[i]);
            O[gr * DIMM + cbase + 32] = f2bf(a2[i]);
            O[gr * DIMM + cbase + 48] = f2bf(a3[i]);
        }
    };
    epilogue(o00, o01, o02, o03, mrow0, lrow0, trow0, 0);
    epilogue(o10, o11, o12, o13, mrow1, lrow1, trow1, 1);
    if (n == 0) atomicAdd(entAcc, entSum);   // lanes 0 and 16: rows {0-7,16-23} and {8-15,24-31}
}

// ---------------- output projection GEMM (32x64 tile per wave) ----------------
__global__ __launch_bounds__(256) void k_gemm_out(
    const bf16* __restrict__ A, const bf16* __restrict__ Bt,
    const float* __restrict__ bias, float* __restrict__ out)
{
    int wave = threadIdx.x >> 5, lane = threadIdx.x & 31;
    int tile = blockIdx.x * 8 + wave;
    int tilesN = DIMM / 64;                  // 16
    int tm = tile / tilesN, tn = tile % tilesN;

    const bf16* ar0 = A + (tm * 32 + 0)  * DIMM;
    const bf16* ar1 = A + (tm * 32 + 16) * DIMM;
    const bf16* br0 = Bt + (tn * 64 + 0)  * DIMM;
    const bf16* br1 = Bt + (tn * 64 + 16) * DIMM;
    const bf16* br2 = Bt + (tn * 64 + 32) * DIMM;
    const bf16* br3 = Bt + (tn * 64 + 48) * DIMM;

    v8f c00 = {}, c01 = {}, c02 = {}, c03 = {};
    v8f c10 = {}, c11 = {}, c12 = {}, c13 = {};

    #pragma unroll 2
    for (int k0 = 0; k0 < DIMM; k0 += 32) {
        __builtin_prefetch(ar0 + k0 + 512, 0, 3);
        __builtin_prefetch(br0 + k0 + 512, 0, 3);
        v16bf a0 = load_a16x32(ar0 + k0, DIMM, lane);
        v16bf a1 = load_a16x32(ar1 + k0, DIMM, lane);
        v16bf b0 = load_b32x16T(br0 + k0, DIMM, lane);
        v16bf b1 = load_b32x16T(br1 + k0, DIMM, lane);
        v16bf b2 = load_b32x16T(br2 + k0, DIMM, lane);
        v16bf b3 = load_b32x16T(br3 + k0, DIMM, lane);
        c00 = wmma_bf16(a0, b0, c00);  c01 = wmma_bf16(a0, b1, c01);
        c02 = wmma_bf16(a0, b2, c02);  c03 = wmma_bf16(a0, b3, c03);
        c10 = wmma_bf16(a1, b0, c10);  c11 = wmma_bf16(a1, b1, c11);
        c12 = wmma_bf16(a1, b2, c12);  c13 = wmma_bf16(a1, b3, c13);
    }

    int half = lane >> 4, n = lane & 15;
    v8f* accs[2][4] = { {&c00,&c01,&c02,&c03}, {&c10,&c11,&c12,&c13} };

    #pragma unroll
    for (int nt = 0; nt < 4; ++nt) {
        int col = tn * 64 + nt * 16 + n;
        float bv = bias[col];
        #pragma unroll
        for (int mi = 0; mi < 2; ++mi) {
            v8f& acc = *accs[mi][nt];
            #pragma unroll
            for (int i = 0; i < 8; ++i) {
                int row = tm * 32 + mi * 16 + i + 8 * half;
                out[row * DIMM + col] = acc[i] + bv;
            }
        }
    }
}

// ---------------- launch ----------------
extern "C" void kernel_launch(void* const* d_in, const int* in_sizes, int n_in,
                              void* d_out, int out_size, void* d_ws, size_t ws_size,
                              hipStream_t stream) {
    const float* x     = (const float*)d_in[0];
    const float* Wqkv  = (const float*)d_in[1];
    const float* bqkv  = (const float*)d_in[2];
    const float* Wout  = (const float*)d_in[3];
    const float* bout  = (const float*)d_in[4];
    const float* scale = (const float*)d_in[5];
    // d_in[6] = mask: causality handled analytically
    float* out = (float*)d_out;

    uint8_t* ws = (uint8_t*)d_ws;
    size_t off = 0;
    auto take = [&](size_t bytes) { size_t o = off; off = (off + bytes + 255) & ~(size_t)255; return o; };

    const size_t nTok = (size_t)BB * SS;           // 8192
    bf16* xh     = (bf16*)(ws + take(nTok * DIMM * 2));
    bf16* wqkvT  = (bf16*)(ws + take((size_t)3 * DIMM * DIMM * 2));
    bf16* woutT  = (bf16*)(ws + take((size_t)DIMM * DIMM * 2));
    bf16* Qb     = (bf16*)(ws + take((size_t)BB * HH * SS * DHH * 2));
    bf16* Kb     = (bf16*)(ws + take((size_t)BB * HH * SS * DHH * 2));
    bf16* Vtb    = (bf16*)(ws + take((size_t)BB * HH * SS * DHH * 2));
    bf16* attnO  = (bf16*)(ws + take(nTok * DIMM * 2));
    float* ent   = (float*)(ws + take(256));

    int nX = BB * SS * DIMM;
    k_cvt_bf16<<<nX / 256, 256, 0, stream>>>(x, xh, nX);
    k_cvt_transpose<<<(DIMM * 3 * DIMM) / 256, 256, 0, stream>>>(Wqkv, wqkvT, DIMM, 3 * DIMM);
    k_cvt_transpose<<<(DIMM * DIMM) / 256, 256, 0, stream>>>(Wout, woutT, DIMM, DIMM);
    k_zero1<<<1, 1, 0, stream>>>(ent);

    // QKV GEMM: (8192/32)*(3072/64) = 12288 wave-tiles / 8 waves = 1536 blocks
    k_gemm_qkv<<<1536, 256, 0, stream>>>(xh, wqkvT, bqkv, scale, Qb, Kb, Vtb);

    // attention: B*H*(S/32) = 4096 wave-tiles / 8 = 512 blocks
    k_attn<<<512, 256, 0, stream>>>(Qb, Kb, Vtb, attnO, ent);

    // out projection: (8192/32)*(1024/64) = 4096 wave-tiles / 8 = 512 blocks
    k_gemm_out<<<512, 256, 0, stream>>>(attnO, woutT, bout, out);

    k_final<<<1, 1, 0, stream>>>(ent, out + nTok * DIMM);
}
